// PointNetpp_21586505630013
// MI455X (gfx1250) — compile-verified, matrix operand order checked by reading, not run on hardware
//
#include <hip/hip_runtime.h>

typedef __attribute__((ext_vector_type(16))) __bf16 v16bf;
typedef __attribute__((ext_vector_type(8)))  __bf16 v8bf;
typedef __attribute__((ext_vector_type(8)))  float  v8f;

#define NSEG  16
#define SEGN  2048
#define SPTS  1024
#define TOTS  16384      // NSEG*SPTS
#define NPT   32768      // total input points
#define NSAMP 32
#define RAD2  1.0f
#define EPS_BN 1e-5f
#define MTOT  524288L    // TOTS*NSAMP

// ---- d_out offsets (in 4-byte elements) ----
#define OUT_XY   0          // 2*16384
#define OUT_XYZ  32768      // 3*16384
#define OUT_PTS  81920      // 256*16384
#define OUT_SPL  4276224    // 17 ints

// ---- workspace byte offsets ----
#define WS_A     0UL                  // bf16 activations [M,128] max  (128 MB)
#define WS_H     134217728UL          // f32 pre-BN h     [M,256] max  (512 MB)
#define WS_STATS 671088640UL          // f32 [16][2][256]
#define WS_AB    671121408UL          // f32 [16][256][2]
#define WS_NXYZ  671154176UL          // f32 [16384][3]

// =================== FPS: one workgroup per segment ===================
__global__ __launch_bounds__(256) void k_fps(const float* __restrict__ xy,
                                             const float* __restrict__ xyz,
                                             float* __restrict__ nxyz,
                                             float* __restrict__ out)
{
  __shared__ float sx[SEGN * 3];
  __shared__ float rv[256];
  __shared__ int   ri[256];
  __shared__ int   sSel;
  const int seg = blockIdx.x;
  const int tid = threadIdx.x;
  for (int i = tid; i < SEGN; i += 256) {
    sx[i*3+0] = xyz[0*NPT + seg*SEGN + i];
    sx[i*3+1] = xyz[1*NPT + seg*SEGN + i];
    sx[i*3+2] = xyz[2*NPT + seg*SEGN + i];
  }
  __syncthreads();
  float dist[8];
  #pragma unroll
  for (int u = 0; u < 8; ++u) dist[u] = 1e10f;
  int sel = 0;
  for (int j = 0; j < SPTS; ++j) {
    const float cx = sx[sel*3+0], cy = sx[sel*3+1], cz = sx[sel*3+2];
    if (tid == 0) {                       // record centroid (before update, like scan)
      const int o = seg*SPTS + j;
      out[OUT_XY  + 0*TOTS + o] = xy[0*NPT + seg*SEGN + sel];
      out[OUT_XY  + 1*TOTS + o] = xy[1*NPT + seg*SEGN + sel];
      out[OUT_XYZ + 0*TOTS + o] = cx;
      out[OUT_XYZ + 1*TOTS + o] = cy;
      out[OUT_XYZ + 2*TOTS + o] = cz;
      nxyz[o*3+0] = cx; nxyz[o*3+1] = cy; nxyz[o*3+2] = cz;
    }
    float bv = -1.f; int bi = SEGN;
    #pragma unroll
    for (int u = 0; u < 8; ++u) {
      const int p = u*256 + tid;
      const float dx = sx[p*3]-cx, dy = sx[p*3+1]-cy, dz = sx[p*3+2]-cz;
      const float d = dx*dx + dy*dy + dz*dz;
      dist[u] = fminf(dist[u], d);
      if (dist[u] > bv || (dist[u] == bv && p < bi)) { bv = dist[u]; bi = p; }
    }
    rv[tid] = bv; ri[tid] = bi;
    __syncthreads();
    for (int s2 = 128; s2 > 0; s2 >>= 1) {      // argmax, first-index tie-break
      if (tid < s2) {
        const float ov = rv[tid+s2]; const int oi = ri[tid+s2];
        if (ov > rv[tid] || (ov == rv[tid] && oi < ri[tid])) { rv[tid] = ov; ri[tid] = oi; }
      }
      __syncthreads();
    }
    if (tid == 0) sSel = ri[0];
    __syncthreads();
    sel = sSel;
  }
}

// ============ Ball query + group: one wave32 per centroid ============
__global__ __launch_bounds__(256) void k_group(const float* __restrict__ xyz,
                                               const float* __restrict__ pts,
                                               const float* __restrict__ nxyz,
                                               __bf16* __restrict__ A0)
{
  __shared__ int gi[8][NSAMP];
  const int tid = threadIdx.x;
  const int w = tid >> 5, lane = tid & 31;
  const int cg  = blockIdx.x*8 + w;          // global centroid id
  const int seg = cg >> 10;
  const float cx = nxyz[cg*3+0], cy = nxyz[cg*3+1], cz = nxyz[cg*3+2];
  const float* bx = xyz + 0*NPT + seg*SEGN;
  const float* by = xyz + 1*NPT + seg*SEGN;
  const float* bz = xyz + 2*NPT + seg*SEGN;
  int count = 0;
  for (int base = 0; base < SEGN && count < NSAMP; base += 32) {
    const int p = base + lane;
    const float dx = bx[p]-cx, dy = by[p]-cy, dz = bz[p]-cz;
    const bool in = (dx*dx + dy*dy + dz*dz) <= RAD2;
    const unsigned mask = (unsigned)__ballot(in ? 1 : 0);    // wave32: low 32 bits
    const int pos = count + __popc(mask & ((1u << lane) - 1u));
    if (in && pos < NSAMP) gi[w][pos] = p;
    count += __popc(mask);
  }
  __syncthreads();
  const int cnt = count < NSAMP ? count : NSAMP;             // cnt >= 1 (centroid)
  const int g0 = gi[w][0];
  if (lane >= cnt) gi[w][lane] = g0;                          // pad with first hit
  __syncthreads();
  // gather: lane j builds one 64-channel bf16 row (3 rel-xyz + 61 feats)
  const int g = gi[w][lane];
  const long row = (long)cg*NSAMP + lane;
  __bf16* dst = A0 + row*64;
  __attribute__((aligned(16))) __bf16 buf[64];
  buf[0] = (__bf16)(bx[g]-cx);
  buf[1] = (__bf16)(by[g]-cy);
  buf[2] = (__bf16)(bz[g]-cz);
  #pragma unroll
  for (int ch = 0; ch < 61; ++ch)
    buf[3+ch] = (__bf16)pts[(long)ch*NPT + seg*SEGN + g];
  #pragma unroll
  for (int q = 0; q < 8; ++q) ((float4*)dst)[q] = ((const float4*)buf)[q];
}

// ===== GEMM (bf16 WMMA, f32 acc) + bias + per-segment BN partials =====
template<int K, int N>
__global__ __launch_bounds__(256) void k_gemm(const __bf16* __restrict__ A,
                                              const float* __restrict__ W,
                                              const float* __restrict__ bias,
                                              float* __restrict__ H,
                                              float* __restrict__ stats)
{
  constexpr int KT = K/32, CT = N/16;
  __shared__ __bf16 wt[KT*CT*512];           // B operand, WMMA lane layout
  __shared__ float ssum[N], ssq[N];
  const int tid = threadIdx.x;
  // Pre-swizzle W (f32 -> bf16) into the 16-bit B-matrix VGPR layout.
  for (int e = tid; e < K*N; e += 256) {
    int t = e;
    const int el = t & 15;  t >>= 4;
    const int ln = t & 31;  t >>= 5;
    const int ct = t % CT, kt = t / CT;
    const int n = ct*16 + (ln & 15);
    const int k = kt*32 + ((ln >> 4) << 3) + (el & 7) + ((el >> 3) << 4);
    wt[e] = (__bf16)W[k*N + n];
  }
  if (tid < N) { ssum[tid] = 0.f; ssq[tid] = 0.f; }
  __syncthreads();

  const int wv = tid >> 5, lane = tid & 31;
  const long rowBase = (long)blockIdx.x*128 + wv*16;   // 128 rows per block
  const int seg = blockIdx.x >> 8;                     // 256 blocks per segment
  const int m = lane & 15, hi = lane >> 4;
  const __bf16* arow = A + (rowBase + m)*K;
  v16bf a[KT];                                         // A: lanes0-15 K0-7/16-23, lanes16-31 K8-15/24-31
  #pragma unroll
  for (int kt = 0; kt < KT; ++kt) {
    const v8bf lo = *(const v8bf*)(arow + kt*32 + hi*8);
    const v8bf hh = *(const v8bf*)(arow + kt*32 + 16 + hi*8);
    a[kt] = __builtin_shufflevector(lo, hh, 0,1,2,3,4,5,6,7,8,9,10,11,12,13,14,15);
  }
  for (int ct = 0; ct < CT; ++ct) {
    v8f acc = {0.f,0.f,0.f,0.f,0.f,0.f,0.f,0.f};
    #pragma unroll
    for (int kt = 0; kt < KT; ++kt) {
      const v16bf b = *(const v16bf*)(wt + ((kt*CT + ct)*32 + lane)*16);
      acc = __builtin_amdgcn_wmma_f32_16x16x32_bf16(false, a[kt], false, b,
                                                    (short)0, acc, false, false);
    }
    const int n = ct*16 + m;
    const float bs = bias[n];
    float ls = 0.f, lq = 0.f;
    const long r0 = rowBase + hi*8;
    #pragma unroll
    for (int i = 0; i < 8; ++i) {
      const float h = acc[i] + bs;
      H[(r0 + i)*N + n] = h;
      ls += h; lq += h*h;
    }
    atomicAdd(&ssum[n], ls);                 // ds_add_f32
    atomicAdd(&ssq[n],  lq);
  }
  __syncthreads();
  if (tid < N) {                             // one global atomic per column per block
    atomicAdd(&stats[(seg*2+0)*N + tid], ssum[tid]);
    atomicAdd(&stats[(seg*2+1)*N + tid], ssq[tid]);
  }
}

// ============= BN finalize: (a,b) per (segment, column) =============
__global__ void k_bnfin(const float* __restrict__ stats, const float* __restrict__ g,
                        const float* __restrict__ be, float* __restrict__ ab, int N)
{
  const int i = blockIdx.x*blockDim.x + threadIdx.x;
  if (i >= NSEG*N) return;
  const int seg = i / N, n = i % N;
  const float inv = 1.f/32768.f;             // S*ns rows per segment
  const float mu  = stats[(seg*2+0)*N + n]*inv;
  const float var = stats[(seg*2+1)*N + n]*inv - mu*mu;
  const float aa  = g[n] * rsqrtf(var + EPS_BN);
  ab[i*2+0] = aa;
  ab[i*2+1] = be[n] - mu*aa;
}

// ========== BN + ReLU + f32->bf16 (feeds next layer's A) ==========
template<int N>
__global__ __launch_bounds__(256) void k_bnact(const float* __restrict__ H,
                                               const float* __restrict__ ab,
                                               __bf16* __restrict__ Ao)
{
  const long total4 = MTOT*(long)N/4;
  const long i = (long)blockIdx.x*256 + threadIdx.x;
  if (i >= total4) return;
  const long b = i*4;
  const long row = b / N; const int col = (int)(b % N);
  const int seg = (int)(row >> 15);
  const float4 h = *(const float4*)(H + b);
  const float* p = ab + ((long)seg*N + col)*2;
  const float v0 = fmaxf(fmaf(p[0], h.x, p[1]), 0.f);
  const float v1 = fmaxf(fmaf(p[2], h.y, p[3]), 0.f);
  const float v2 = fmaxf(fmaf(p[4], h.z, p[5]), 0.f);
  const float v3 = fmaxf(fmaf(p[6], h.w, p[7]), 0.f);
  __attribute__((aligned(8))) __bf16 o[4];
  o[0] = (__bf16)v0; o[1] = (__bf16)v1; o[2] = (__bf16)v2; o[3] = (__bf16)v3;
  *(uint2*)(Ao + b) = *(const uint2*)o;
}

// ===== layer 3: BN + ReLU fused into neighbor max-pool -> d_out =====
__global__ __launch_bounds__(256) void k_maxpool(const float* __restrict__ H,
                                                 const float* __restrict__ ab,
                                                 float* __restrict__ outPts)
{
  const int sg = blockIdx.x;                 // 0..16383
  const int n  = threadIdx.x;                // 0..255
  const int seg = sg >> 10;
  const float* p = ab + ((long)seg*256 + n)*2;
  const float aa = p[0], bb = p[1];
  const float* base = H + (long)sg*NSAMP*256 + n;
  float m = -3.4e38f;
  #pragma unroll 8
  for (int j = 0; j < NSAMP; ++j) m = fmaxf(m, fmaf(aa, base[(long)j*256], bb));
  outPts[(long)n*TOTS + sg] = fmaxf(m, 0.f); // relu(max) == max(relu)
}

__global__ void k_split(float* outTail)
{
  const int t = threadIdx.x;
  if (t < NSEG + 1) ((int*)outTail)[t] = t*SPTS;
}

extern "C" void kernel_launch(void* const* d_in, const int* in_sizes, int n_in,
                              void* d_out, int out_size, void* d_ws, size_t ws_size,
                              hipStream_t stream) {
  (void)in_sizes; (void)n_in; (void)out_size; (void)ws_size;
  const float* xy   = (const float*)d_in[0];
  const float* xyz  = (const float*)d_in[1];
  const float* pts  = (const float*)d_in[2];
  const float* w1 = (const float*)d_in[4],  *c1 = (const float*)d_in[5];
  const float* g1 = (const float*)d_in[6],  *be1 = (const float*)d_in[7];
  const float* w2 = (const float*)d_in[8],  *c2 = (const float*)d_in[9];
  const float* g2 = (const float*)d_in[10], *be2 = (const float*)d_in[11];
  const float* w3 = (const float*)d_in[12], *c3 = (const float*)d_in[13];
  const float* g3 = (const float*)d_in[14], *be3 = (const float*)d_in[15];
  float* out = (float*)d_out;
  char* ws = (char*)d_ws;
  __bf16* A     = (__bf16*)(ws + WS_A);
  float*  H     = (float*) (ws + WS_H);
  float*  stats = (float*) (ws + WS_STATS);
  float*  ab    = (float*) (ws + WS_AB);
  float*  nxyz  = (float*) (ws + WS_NXYZ);

  k_fps  <<<NSEG, 256, 0, stream>>>(xy, xyz, nxyz, out);
  k_group<<<TOTS/8, 256, 0, stream>>>(xyz, pts, nxyz, A);

  // layer 1: 64 -> 64
  hipMemsetAsync(stats, 0, NSEG*2*64*sizeof(float), stream);
  k_gemm<64,64><<<4096, 256, 0, stream>>>(A, w1, c1, H, stats);
  k_bnfin<<<(NSEG*64 + 255)/256, 256, 0, stream>>>(stats, g1, be1, ab, 64);
  k_bnact<64><<<(int)((MTOT*64/4 + 255)/256), 256, 0, stream>>>(H, ab, A);

  // layer 2: 64 -> 128
  hipMemsetAsync(stats, 0, NSEG*2*128*sizeof(float), stream);
  k_gemm<64,128><<<4096, 256, 0, stream>>>(A, w2, c2, H, stats);
  k_bnfin<<<(NSEG*128 + 255)/256, 256, 0, stream>>>(stats, g2, be2, ab, 128);
  k_bnact<128><<<(int)((MTOT*128/4 + 255)/256), 256, 0, stream>>>(H, ab, A);

  // layer 3: 128 -> 256, fused BN+ReLU+maxpool
  hipMemsetAsync(stats, 0, NSEG*2*256*sizeof(float), stream);
  k_gemm<128,256><<<4096, 256, 0, stream>>>(A, w3, c3, H, stats);
  k_bnfin<<<(NSEG*256 + 255)/256, 256, 0, stream>>>(stats, g3, be3, ab, 256);
  k_maxpool<<<TOTS, 256, 0, stream>>>(H, ab, out + OUT_PTS);

  k_split<<<1, 32, 0, stream>>>(out + OUT_SPL);
}
